// VanillaLSTMlayer_59184649339321
// MI455X (gfx1250) — compile-verified, matrix operand order checked by reading, not run on hardware
//
#include <hip/hip_runtime.h>
#include <hip/hip_bf16.h>

// ---------------- problem constants ----------------
static constexpr int Bx   = 128;   // batch
static constexpr int Sx   = 512;   // sequence length
static constexpr int Hd   = 512;   // hidden
static constexpr int NWG  = 16;    // workgroups (persistent)
static constexpr int COLS = 32;    // hidden columns owned per WG (16*32 = 512)
static constexpr int NTHR = 512;   // 16 waves of 32
static constexpr int KP   = 520;   // padded K stride in elements (1040B: bank-conflict-free, 16B aligned)

// ---------------- LDS memory map (bytes) ----------------
static constexpr int OFF_W   = 0;
static constexpr int SZ_W    = 4 * COLS * KP * 2;        // 133120 : bf16 weights [gate][col][k]
static constexpr int OFF_HA  = OFF_W + SZ_W;             // 133120
static constexpr int SZ_HA   = Bx * KP * 2;              // 133120 : bf16 H staging [m][k]
static constexpr int OFF_G   = OFF_HA;                   // aliased: f32 gates [4][Bx][COLS] = 65536
static constexpr int OFF_C   = OFF_HA + SZ_HA;           // 266240 : f32 C state [Bx][COLS]
static constexpr int SZ_C    = Bx * COLS * 4;            // 16384
static constexpr int OFF_BI  = OFF_C + SZ_C;             // 282624 : f32 bias [4][COLS]
static constexpr int OFF_WX  = OFF_BI + 4 * COLS * 4;    // 283136 : f32 x-weight row [4][COLS]
static constexpr int LDS_BYTES = OFF_WX + 4 * COLS * 4;  // 283648 (< 320KB WGP LDS)

typedef __attribute__((ext_vector_type(16))) __bf16    v16bf;
typedef __attribute__((ext_vector_type(8)))  float     v8f;
typedef __attribute__((ext_vector_type(4)))  unsigned  u32x4;
typedef __attribute__((ext_vector_type(4)))  int       v4i;

union FragBF16 { v16bf v; u32x4 q[2]; };

// CDNA5 async global->LDS copy path (ASYNCcnt), if the toolchain exposes it.
#if defined(__has_builtin)
#if __has_builtin(__builtin_amdgcn_global_load_async_to_lds_b128) && \
    __has_builtin(__builtin_amdgcn_s_wait_asynccnt)
#define USE_ASYNC_LDS 1
#endif
#endif
#ifndef USE_ASYNC_LDS
#define USE_ASYNC_LDS 0
#endif

#if USE_ASYNC_LDS
typedef __attribute__((address_space(1))) v4i* gp128;    // global 128-bit chunk
typedef __attribute__((address_space(3))) v4i* lp128;    // LDS 128-bit chunk
#endif

__device__ __forceinline__ unsigned short f2bf(float f) {
    unsigned u = __float_as_uint(f);
    unsigned r = 0x7FFFu + ((u >> 16) & 1u);   // round-to-nearest-even
    return (unsigned short)((u + r) >> 16);
}
__device__ __forceinline__ float fast_sigmoid(float v) {
    return 1.0f / (1.0f + __expf(-v));
}
__device__ __forceinline__ float fast_tanh(float v) {
    float e = __expf(2.0f * v);
    return (e - 1.0f) / (e + 1.0f);
}

__device__ __forceinline__ void grid_barrier(unsigned* bar, unsigned* gen, unsigned nwg) {
    __threadfence();           // publish this WG's H stores (device scope release)
    __syncthreads();
    if (threadIdx.x == 0) {
        unsigned g = __hip_atomic_load(gen, __ATOMIC_ACQUIRE, __HIP_MEMORY_SCOPE_AGENT);
        unsigned a = __hip_atomic_fetch_add(bar, 1u, __ATOMIC_ACQ_REL, __HIP_MEMORY_SCOPE_AGENT) + 1u;
        if (a == nwg) {
            __hip_atomic_store(bar, 0u, __ATOMIC_RELAXED, __HIP_MEMORY_SCOPE_AGENT);
            __hip_atomic_fetch_add(gen, 1u, __ATOMIC_RELEASE, __HIP_MEMORY_SCOPE_AGENT);
        } else {
            while (__hip_atomic_load(gen, __ATOMIC_ACQUIRE, __HIP_MEMORY_SCOPE_AGENT) == g) {
                __builtin_amdgcn_s_sleep(2);
            }
        }
    }
    __syncthreads();
    __threadfence();           // acquire: invalidate vector caches in every wave/CU
}

// ---------------- init: zero barrier state, H(0) -> bf16 ping buffer ----------------
extern "C" __global__ void lstm_init(const float* __restrict__ Hprev,
                                     unsigned short* __restrict__ Hb0,
                                     unsigned* __restrict__ syncv) {
    int i = blockIdx.x * blockDim.x + threadIdx.x;
    if (i < 2) syncv[i] = 0u;
    for (int e = i; e < Bx * Hd; e += gridDim.x * blockDim.x)
        Hb0[e] = f2bf(Hprev[e]);
}

// ---------------- persistent LSTM kernel ----------------
extern "C" __global__ __launch_bounds__(NTHR)
void lstm_persistent(const float* __restrict__ x,
                     const float* __restrict__ Cprev,
                     const float* __restrict__ Wi, const float* __restrict__ bi,
                     const float* __restrict__ Wf, const float* __restrict__ bfv,
                     const float* __restrict__ Wc, const float* __restrict__ bc,
                     const float* __restrict__ Wo, const float* __restrict__ bo,
                     unsigned short* __restrict__ Hbuf0,
                     unsigned short* __restrict__ Hbuf1,
                     unsigned* __restrict__ bar, unsigned* __restrict__ gen,
                     float* __restrict__ out)
{
    extern __shared__ char smem[];
    unsigned short* Wlds = (unsigned short*)(smem + OFF_W);
    unsigned short* HA   = (unsigned short*)(smem + OFF_HA);
    float*          G    = (float*)(smem + OFF_G);
    float*          Cs   = (float*)(smem + OFF_C);
    float*          Bs   = (float*)(smem + OFF_BI);
    float*          WX   = (float*)(smem + OFF_WX);

    const int tid   = threadIdx.x;
    const int wg    = blockIdx.x;
    const int nBase = wg * COLS;

    const float* Wg[4] = {Wi, Wf, Wc, Wo};
    const float* Bg[4] = {bi, bfv, bc, bo};

    // ---- one-time: weight slice -> LDS bf16 ([gate][col][k], k-stride KP) ----
#pragma unroll
    for (int g = 0; g < 4; ++g) {
        const float* W = Wg[g];
        for (int e = tid; e < Hd * COLS; e += NTHR) {
            int k = e >> 5;          // 0..511
            int c = e & 31;
            Wlds[(g * COLS + c) * KP + k] = f2bf(W[(k + 1) * Hd + nBase + c]);
        }
    }
    if (tid < 4 * COLS) {
        int g = tid >> 5, c = tid & 31;
#pragma unroll
        for (int gg = 0; gg < 4; ++gg) {
            if (gg == g) {
                WX[tid] = Wg[gg][nBase + c];          // row 0 of W = x weight
                Bs[tid] = Bg[gg][nBase + c];
            }
        }
    }
    // ---- one-time: C state -> LDS ----
    for (int e = tid; e < Bx * COLS; e += NTHR) {
        int m = e >> 5, c = e & 31;
        Cs[e] = Cprev[m * Hd + nBase + c];
    }
    __syncthreads();

    const int wave = tid >> 5;
    const int lane = tid & 31;
    const int half = lane >> 4;
    const int lr   = lane & 15;
    const int m0   = (wave & 7) * 16;       // M tile base
    const int ttB  = (wave >> 3) * 4;       // first of 4 N-tiles (tile = gate*2 + coltile)

    for (int t = 0; t < Sx; ++t) {
        const unsigned short* Hcur = (t & 1) ? Hbuf1 : Hbuf0;
        unsigned short*       Hnxt = (t & 1) ? Hbuf0 : Hbuf1;

        // ---- stage H(t) [Bx][Hd] bf16 into LDS with padded stride ----
        {
            int r = tid >> 2, qq = tid & 3;
#if USE_ASYNC_LDS
            // CDNA5 async copy engine: global -> LDS without touching VGPRs.
            const char* src = (const char*)(Hcur + r * Hd + qq * 128);
            char*       dst = (char*)HA + (r * KP + qq * 128) * 2;
#pragma unroll
            for (int i = 0; i < 16; ++i) {
                __builtin_amdgcn_global_load_async_to_lds_b128(
                    (gp128)(src + i * 16), (lp128)(dst + i * 16), 0, 0);
            }
            __builtin_amdgcn_s_wait_asynccnt(0);
#else
            const u32x4* src = (const u32x4*)(Hcur + r * Hd + qq * 128);
            u32x4* dst = (u32x4*)((char*)HA + (r * KP + qq * 128) * 2);
#pragma unroll
            for (int i = 0; i < 16; ++i) dst[i] = src[i];
#endif
        }
        __syncthreads();

        // ---- GEMM: 4 output tiles per wave, K-loop of 16 x wmma bf16 k=32 ----
        v8f acc[4] = {};
        const unsigned short* Arow = HA + (m0 + lr) * KP;
        const unsigned short* Brow[4];
#pragma unroll
        for (int i = 0; i < 4; ++i) {
            int tt = ttB + i;
            int g  = tt >> 1;
            int c0 = (tt & 1) * 16;
            Brow[i] = Wlds + (g * COLS + c0 + lr) * KP;
        }
#pragma unroll 4
        for (int kk = 0; kk < 16; ++kk) {
            int kb = kk * 32 + half * 8;     // ISA lane/K split for 16-bit A/B
            FragBF16 a;
            a.q[0] = *(const u32x4*)(Arow + kb);
            a.q[1] = *(const u32x4*)(Arow + kb + 16);
#pragma unroll
            for (int i = 0; i < 4; ++i) {
                FragBF16 b;
                b.q[0] = *(const u32x4*)(Brow[i] + kb);
                b.q[1] = *(const u32x4*)(Brow[i] + kb + 16);
                acc[i] = __builtin_amdgcn_wmma_f32_16x16x32_bf16(
                    false, a.v, false, b.v, (short)0, acc[i], false, false);
            }
        }
        __syncthreads();   // everyone done reading HA before we alias it as G

        // ---- scatter accumulators to LDS gate buffer G[g][m][c] ----
#pragma unroll
        for (int i = 0; i < 4; ++i) {
            int tt = ttB + i;
            int g  = tt >> 1;
            int c0 = (tt & 1) * 16;
#pragma unroll
            for (int p = 0; p < 8; ++p) {
                int m = m0 + p + 8 * half;
                G[(g * Bx + m) * COLS + c0 + lr] = acc[i][p];
            }
        }
        __syncthreads();

        // ---- elementwise LSTM update (8 elements / thread) ----
#pragma unroll
        for (int i = 0; i < 8; ++i) {
            int idx = tid + NTHR * i;          // 0..4095
            int m = idx >> 5, c = idx & 31;
            float xv = x[m * Sx + t];
            if (i == 0 && t + 1 < Sx) __builtin_prefetch(&x[m * Sx + t + 1], 0, 1);
            float Iv = G[(0 * Bx + m) * COLS + c] + xv * WX[0 * COLS + c] + Bs[0 * COLS + c];
            float Fv = G[(1 * Bx + m) * COLS + c] + xv * WX[1 * COLS + c] + Bs[1 * COLS + c];
            float Cv = G[(2 * Bx + m) * COLS + c] + xv * WX[2 * COLS + c] + Bs[2 * COLS + c];
            float Ov = G[(3 * Bx + m) * COLS + c] + xv * WX[3 * COLS + c] + Bs[3 * COLS + c];
            Iv = fast_sigmoid(Iv);
            Fv = fast_sigmoid(Fv);
            Cv = fast_tanh(Cv);
            float Cn = Fv * Cs[idx] + Iv * Cv;
            Cs[idx] = Cn;
            float Hn = Ov * fast_tanh(Cn);     // reference applies no sigmoid to O
            Hnxt[m * Hd + nBase + c] = f2bf(Hn);
            if (t == Sx - 1) {
                int o = m * Hd + nBase + c;
                out[o]                = Hn;    // hidden_sequence (== H_t)
                out[Bx * Hd + o]      = Hn;    // H_t
                out[2 * Bx * Hd + o]  = Cn;    // C_t
            }
        }

        if (t != Sx - 1) grid_barrier(bar, gen, NWG);
    }
}

// ---------------- host entry ----------------
extern "C" void kernel_launch(void* const* d_in, const int* in_sizes, int n_in,
                              void* d_out, int out_size, void* d_ws, size_t ws_size,
                              hipStream_t stream) {
    (void)in_sizes; (void)n_in; (void)out_size; (void)ws_size;
    const float* x     = (const float*)d_in[0];
    const float* Hprev = (const float*)d_in[1];
    const float* Cprev = (const float*)d_in[2];
    const float* Wi    = (const float*)d_in[3];
    const float* bi    = (const float*)d_in[4];
    const float* Wf    = (const float*)d_in[5];
    const float* bfv   = (const float*)d_in[6];
    const float* Wc    = (const float*)d_in[7];
    const float* bc    = (const float*)d_in[8];
    const float* Wo    = (const float*)d_in[9];
    const float* bo    = (const float*)d_in[10];
    float* out = (float*)d_out;

    unsigned*       syncv = (unsigned*)d_ws;
    unsigned short* Hb0   = (unsigned short*)((char*)d_ws + 256);
    unsigned short* Hb1   = Hb0 + Bx * Hd;

    // allow >64KB dynamic LDS (idempotent; ignore errors)
    (void)hipFuncSetAttribute(reinterpret_cast<const void*>(lstm_persistent),
                              hipFuncAttributeMaxDynamicSharedMemorySize, LDS_BYTES);

    lstm_init<<<64, 256, 0, stream>>>(Hprev, Hb0, syncv);
    lstm_persistent<<<NWG, NTHR, LDS_BYTES, stream>>>(
        x, Cprev, Wi, bi, Wf, bfv, Wc, bc, Wo, bo,
        Hb0, Hb1, &syncv[0], &syncv[1], out);
}
